// DynamicHead_64209761076026
// MI455X (gfx1250) — compile-verified, matrix operand order checked by reading, not run on hardware
//
#include <hip/hip_runtime.h>
#include <hip/hip_bf16.h>
#include <math.h>

// ---------------------------------------------------------------------------
// DynamicHead forward for MI455X (gfx1250, wave32, WMMA).
// All matmuls run through v_wmma_f32_16x16x32_f16 (f16 inputs converted on the
// fly from f32, f32 accumulation). One generic GEMM kernel covers plain,
// biased, activated, batched and strided-batched cases (incl. reading
// roi_features (P,B,D) directly as a batched (B:,P,D) operand via lda).
// A and B tiles are staged through LDS: coalesced global b128 reads, f16
// conversion once per block, fragments fetched as ds_load_b128.
// ---------------------------------------------------------------------------

typedef __attribute__((ext_vector_type(16))) _Float16 v16h;
typedef __attribute__((ext_vector_type(8)))  _Float16 v8h;
typedef __attribute__((ext_vector_type(4)))  _Float16 v4h;
typedef __attribute__((ext_vector_type(2)))  _Float16 v2h;
typedef __attribute__((ext_vector_type(8)))  float    v8f;

static constexpr int Np  = 8;     // images
static constexpr int NR  = 300;   // proposals
static constexpr int Dm  = 256;   // model dim
static constexpr int Pp  = 49;    // roi cells
static constexpr int DFF = 2048;
static constexpr int NC  = 80;
static constexpr int DDY = 64;
static constexpr int Hh  = 8;
static constexpr int HD  = Dm / Hh;          // 32
static constexpr int Bt  = Np * NR;          // 2400
static constexpr int NH  = Np * Hh;          // 64 batch*head
static constexpr int SBS = 40;               // LDS K-stride (halves): 80B rows, 16B-aligned
#define SCALE_CLAMP 8.740336742730447f       // ln(100000/16)

__device__ inline v16h pack16(float4 a0, float4 a1, float4 a2, float4 a3)
{
  v16h r;
  r[0]  = (_Float16)a0.x; r[1]  = (_Float16)a0.y; r[2]  = (_Float16)a0.z; r[3]  = (_Float16)a0.w;
  r[4]  = (_Float16)a1.x; r[5]  = (_Float16)a1.y; r[6]  = (_Float16)a1.z; r[7]  = (_Float16)a1.w;
  r[8]  = (_Float16)a2.x; r[9]  = (_Float16)a2.y; r[10] = (_Float16)a2.z; r[11] = (_Float16)a2.w;
  r[12] = (_Float16)a3.x; r[13] = (_Float16)a3.y; r[14] = (_Float16)a3.z; r[15] = (_Float16)a3.w;
  return r;
}
__device__ inline v4h pack4(float4 v)
{ v4h r; r[0]=(_Float16)v.x; r[1]=(_Float16)v.y; r[2]=(_Float16)v.z; r[3]=(_Float16)v.w; return r; }
__device__ inline v2h pack2(float a, float b)
{ v2h r; r[0]=(_Float16)a; r[1]=(_Float16)b; return r; }

// Assemble a 16-half fragment from two contiguous 8-half LDS runs.
__device__ inline v16h frag_from_lds(const _Float16* p)
{
  v8h a0 = *(const v8h*)(p);        // ds_load_b128
  v8h a1 = *(const v8h*)(p + 16);   // ds_load_b128
  v16h r;
#pragma unroll
  for (int i = 0; i < 8; ++i) { r[i] = a0[i]; r[8 + i] = a1[i]; }
  return r;
}

// ---------------------------------------------------------------------------
// Generic WMMA GEMM:  C[z] = act( alpha * A[z] @ B[z] + bias )
//   A: (M,K) rows, stride lda, batch stride bsA
//   B: TB=0 -> (K,N) rows stride ldb;  TB=1 -> (N,K) rows stride ldb
//   C: (M,N) rows, stride ldc
// 4 waves / block: same M-tile, 4 adjacent N-tiles (64 columns).
// OOB M/N lanes are CLAMPED, not masked: a WMMA A-lane m only feeds C-row m
// and a B-lane n only feeds C-col n, and OOB rows/cols are never stored, so
// garbage in clamped lanes/columns is harmless. Only K needs zero-fill.
// Requires (guaranteed by all call sites): N % 4 == 0, K % 4 == 0, ldb >= N,
// all row strides multiples of 4 floats, 16B-aligned bases.
// ACT: 0 none, 1 relu, 2 tanh-gelu.
// ---------------------------------------------------------------------------
template<int ACT, int TB>
__global__ __launch_bounds__(128)
void k_gemm(const float* __restrict__ A, const float* __restrict__ Bm,
            const float* __restrict__ bias, float* __restrict__ C,
            int M, int Nn, int K, int lda, int ldb, int ldc,
            long long bsA, long long bsB, long long bsC, float alpha)
{
  const int tid  = threadIdx.x;
  const int wave = tid >> 5;
  const int lane = tid & 31;
  const int tn = blockIdx.x * 4 + wave;   // N-tile index
  const int tm = blockIdx.y;              // M-tile index
  const long long z = blockIdx.z;
  A  += z * bsA;  Bm += z * bsB;  C += z * bsC;

  const int hi    = lane >> 4;          // half-wave select
  const int l15   = lane & 15;
  const int ncol0 = tn * 16 + l15;
  const int kb    = hi * 8;             // ISA 16-bit fragment K sub-offset

  v8f acc = {};
  const int kmain = K & ~31;

  if constexpr (TB) {
    // -------- direct path: B^T rows are K-contiguous -> 4x b128 each -------
    const int mrow = min(tm * 16 + l15, M - 1);
    const int ncol = min(ncol0, Nn - 1);
    const float* __restrict__ arow = A + (long long)mrow * lda;
    const float* __restrict__ brow = Bm + (long long)ncol * ldb;
    int k0 = 0;
    for (; k0 < kmain; k0 += 32) {
      if (k0 + 32 < kmain) {
        __builtin_prefetch(arow + k0 + 32 + kb, 0, 1);
        __builtin_prefetch(brow + k0 + 32 + kb, 0, 1);
      }
      v16h af = pack16(*(const float4*)(arow + k0 + kb),
                       *(const float4*)(arow + k0 + kb + 4),
                       *(const float4*)(arow + k0 + kb + 16),
                       *(const float4*)(arow + k0 + kb + 20));
      v16h bf = pack16(*(const float4*)(brow + k0 + kb),
                       *(const float4*)(brow + k0 + kb + 4),
                       *(const float4*)(brow + k0 + kb + 16),
                       *(const float4*)(brow + k0 + kb + 20));
      acc = __builtin_amdgcn_wmma_f32_16x16x32_f16(false, af, false, bf,
                                                   (short)0, acc, false, false);
    }
    if (k0 < K) {
      v16h af, bf;
#pragma unroll
      for (int i = 0; i < 16; ++i) {
        int k  = k0 + kb + (i < 8 ? i : i + 8);
        int kc = min(k, K - 1);
        float av = arow[kc];
        float bv = brow[kc];
        af[i] = (k < K) ? (_Float16)av : (_Float16)0.f;
        bf[i] = (k < K) ? (_Float16)bv : (_Float16)0.f;
      }
      acc = __builtin_amdgcn_wmma_f32_16x16x32_f16(false, af, false, bf,
                                                   (short)0, acc, false, false);
    }
  } else {
    // ------ LDS-staged path: block stages 16x32 A tile and 32x64 B tile ----
    __shared__ _Float16 sA[16 * SBS];          // row-major:  row*SBS + k
    __shared__ _Float16 sB[64 * SBS];          // col-major:  col*SBS + k
    // A staging: thread -> one float4 (row ar, k quad acq)
    const int ar  = tid >> 3;                  // 0..15
    const int acq = (tid & 7) * 4;             // 0..28
    const float* __restrict__ arow = A + (long long)min(tm * 16 + ar, M - 1) * lda;
    // B staging: thread -> 4 adjacent K rows (kg..kg+3), one col quad
    const int cq = (tid & 15) * 4;             // local column quad
    const int kg = (tid >> 4) * 4;             // 0..28
    const int cg = min(blockIdx.x * 64 + cq, Nn - 4);  // clamped (N%4==0)

    int k0 = 0;
    for (; k0 < kmain; k0 += 32) {
      __syncthreads();
      // ---- stage A: 1x b128 + 1x ds_store_b64 ----
      *(v4h*)&sA[ar * SBS + acq] = pack4(*(const float4*)(arow + k0 + acq));
      // ---- stage B: 4x b128 + 8x ds_store_b32 (K-pairs packed) ----
      float4 r0 = *(const float4*)(Bm + (long long)(k0 + kg + 0) * ldb + cg);
      float4 r1 = *(const float4*)(Bm + (long long)(k0 + kg + 1) * ldb + cg);
      float4 r2 = *(const float4*)(Bm + (long long)(k0 + kg + 2) * ldb + cg);
      float4 r3 = *(const float4*)(Bm + (long long)(k0 + kg + 3) * ldb + cg);
      *(v2h*)&sB[(cq + 0) * SBS + kg]     = pack2(r0.x, r1.x);
      *(v2h*)&sB[(cq + 0) * SBS + kg + 2] = pack2(r2.x, r3.x);
      *(v2h*)&sB[(cq + 1) * SBS + kg]     = pack2(r0.y, r1.y);
      *(v2h*)&sB[(cq + 1) * SBS + kg + 2] = pack2(r2.y, r3.y);
      *(v2h*)&sB[(cq + 2) * SBS + kg]     = pack2(r0.z, r1.z);
      *(v2h*)&sB[(cq + 2) * SBS + kg + 2] = pack2(r2.z, r3.z);
      *(v2h*)&sB[(cq + 3) * SBS + kg]     = pack2(r0.w, r1.w);
      *(v2h*)&sB[(cq + 3) * SBS + kg + 2] = pack2(r2.w, r3.w);
      __syncthreads();
      if (k0 + 32 < kmain) {  // uniform -> global_prefetch_b8
        __builtin_prefetch(arow + k0 + 32 + acq, 0, 1);
        __builtin_prefetch(Bm + (long long)(k0 + 32 + kg) * ldb + cg, 0, 1);
      }
      v16h af = frag_from_lds(&sA[l15 * SBS + kb]);
      v16h bf = frag_from_lds(&sB[(wave * 16 + l15) * SBS + kb]);
      acc = __builtin_amdgcn_wmma_f32_16x16x32_f16(false, af, false, bf,
                                                   (short)0, acc, false, false);
    }
    if (k0 < K) {  // K tail (K=300 path): zero-fill at staging
      __syncthreads();
      {
        v4h ha;
#pragma unroll
        for (int j = 0; j < 4; ++j) {          // scalar clamped loads (rare path)
          int k = k0 + acq + j;
          float v = arow[min(k, K - 1)];
          ha[j] = (k < K) ? (_Float16)v : (_Float16)0.f;
        }
        *(v4h*)&sA[ar * SBS + acq] = ha;
      }
      float4 rr[4];
      bool   ok[4];
#pragma unroll
      for (int j = 0; j < 4; ++j) {            // row-granular clamp is exact
        int k = k0 + kg + j;
        rr[j] = *(const float4*)(Bm + (long long)min(k, K - 1) * ldb + cg);
        ok[j] = (k < K);
      }
      float c[4][4] = {{rr[0].x, rr[0].y, rr[0].z, rr[0].w},
                       {rr[1].x, rr[1].y, rr[1].z, rr[1].w},
                       {rr[2].x, rr[2].y, rr[2].z, rr[2].w},
                       {rr[3].x, rr[3].y, rr[3].z, rr[3].w}};
#pragma unroll
      for (int j = 0; j < 4; ++j)
#pragma unroll
        for (int j2 = 0; j2 < 4; ++j2)
          if (!ok[j]) c[j][j2] = 0.f;
#pragma unroll
      for (int j2 = 0; j2 < 4; ++j2) {
        *(v2h*)&sB[(cq + j2) * SBS + kg]     = pack2(c[0][j2], c[1][j2]);
        *(v2h*)&sB[(cq + j2) * SBS + kg + 2] = pack2(c[2][j2], c[3][j2]);
      }
      __syncthreads();
      v16h af = frag_from_lds(&sA[l15 * SBS + kb]);
      v16h bf = frag_from_lds(&sB[(wave * 16 + l15) * SBS + kb]);
      acc = __builtin_amdgcn_wmma_f32_16x16x32_f16(false, af, false, bf,
                                                   (short)0, acc, false, false);
    }
  }

  if (ncol0 >= Nn) return;
  const float bv = bias ? bias[ncol0] : 0.f;
#pragma unroll
  for (int r = 0; r < 8; ++r) {
    int m = tm * 16 + r + 8 * hi;   // ISA f32 C layout: lane=N, vgpr r -> M
    if (m < M) {
      float v = acc[r] * alpha + bv;
      if (ACT == 1) v = fmaxf(v, 0.f);
      else if (ACT == 2) {
        float t = 0.7978845608028654f * (v + 0.044715f * v * v * v);
        v = 0.5f * v * (1.f + tanhf(t));
      }
      C[(long long)m * ldc + ncol0] = v;
    }
  }
}

// ---------------- LayerNorm (optional residual, optional relu) -------------
template<int W>
__global__ __launch_bounds__(64)
void k_layernorm(const float* __restrict__ X, const float* __restrict__ R,
                 const float* __restrict__ g, const float* __restrict__ bb,
                 float* __restrict__ Y, int relu)
{
  const long long row = blockIdx.x;
  const float* x = X + row * W;
  const float* r = R ? R + row * W : nullptr;
  float*       y = Y + row * W;
  __shared__ float s1[64], s2[64];
  const int t = threadIdx.x;
  constexpr int PER = W / 64;
  float lv[PER];
  float sum = 0.f, sq = 0.f;
#pragma unroll
  for (int i = 0; i < PER; ++i) {
    float v = x[t + i * 64];
    if (r) v += r[t + i * 64];
    lv[i] = v; sum += v; sq += v * v;
  }
  s1[t] = sum; s2[t] = sq; __syncthreads();
  for (int s = 32; s > 0; s >>= 1) {
    if (t < s) { s1[t] += s1[t + s]; s2[t] += s2[t + s]; }
    __syncthreads();
  }
  const float m  = s1[0] / W;
  const float vr = s2[0] / W - m * m;
  const float rs = rsqrtf(vr + 1e-5f);
#pragma unroll
  for (int i = 0; i < PER; ++i) {
    int c = t + i * 64;
    float v = (lv[i] - m) * rs * g[c] + bb[c];
    if (relu) v = fmaxf(v, 0.f);
    y[c] = v;
  }
}

// ---------------- softmax over 300 (in place), one block per row -----------
__global__ __launch_bounds__(128)
void k_softmax300(float* __restrict__ S)
{
  float* p = S + (long long)blockIdx.x * NR;
  __shared__ float red[128];
  const int t = threadIdx.x;
  float mx = -1e30f;
  for (int j = t; j < NR; j += 128) mx = fmaxf(mx, p[j]);
  red[t] = mx; __syncthreads();
  for (int s = 64; s > 0; s >>= 1) { if (t < s) red[t] = fmaxf(red[t], red[t + s]); __syncthreads(); }
  mx = red[0]; __syncthreads();
  float sum = 0.f;
  for (int j = t; j < NR; j += 128) { float e = __expf(p[j] - mx); p[j] = e; sum += e; }
  red[t] = sum; __syncthreads();
  for (int s = 64; s > 0; s >>= 1) { if (t < s) red[t] += red[t + s]; __syncthreads(); }
  const float inv = 1.f / red[0];
  for (int j = t; j < NR; j += 128) p[j] *= inv;
}

// ------------- qkv (B,768) -> Q,K,V as (NH,300,32) each --------------------
__global__ void k_qkv_split(const float* __restrict__ qkv, float* __restrict__ Q,
                            float* __restrict__ Kq, float* __restrict__ V)
{
  int idx = blockIdx.x * blockDim.x + threadIdx.x;
  if (idx >= Bt * Dm) return;
  int row = idx / Dm, c = idx % Dm;
  int b = row / NR, n = row % NR, h = c / HD, d = c % HD;
  long long src = (long long)row * (3 * Dm);
  long long dst = (((long long)(b * Hh + h)) * NR + n) * HD + d;
  Q [dst] = qkv[src + c];
  Kq[dst] = qkv[src + Dm + c];
  V [dst] = qkv[src + 2 * Dm + c];
}

// ------------- O (NH,300,32) -> (B,256) ------------------------------------
__global__ void k_head_merge(const float* __restrict__ O, float* __restrict__ Y)
{
  int idx = blockIdx.x * blockDim.x + threadIdx.x;
  if (idx >= Bt * Dm) return;
  int row = idx / Dm, c = idx % Dm;
  int b = row / NR, n = row % NR, h = c / HD, d = c % HD;
  Y[idx] = O[(((long long)(b * Hh + h)) * NR + n) * HD + d];
}

// ------------- prune-policy gates (log_softmax(2) is monotone) -------------
__global__ void k_policy(const float* __restrict__ h2,
                         const float* __restrict__ dy_w, const float* __restrict__ dy_b,
                         const float* __restrict__ st_w, const float* __restrict__ st_b,
                         const float* __restrict__ policy, const float* __restrict__ policy_st,
                         float* __restrict__ pol, float* __restrict__ ap)
{
  int i = blockIdx.x * blockDim.x + threadIdx.x;
  if (i >= Bt) return;
  const float* h = h2 + (long long)i * DDY;
  float z0 = dy_b[0], z1 = dy_b[1], s0 = st_b[0], s1 = st_b[1];
  for (int k = 0; k < DDY; ++k) {
    float v = h[k];
    z0 += v * dy_w[k * 2];  z1 += v * dy_w[k * 2 + 1];
    s0 += v * st_w[k * 2];  s1 += v * st_w[k * 2 + 1];
  }
  float app = policy[i] + policy_st[i];
  float p   = (z0 > z1) ? app : 0.f;
  float pst = (s0 > s1) ? (1.f - p) * app : 0.f;
  pol[i] = p;
  ap[i]  = p + pst;
}

// ------------- pf3 = pf + fdyn * pol[row] ----------------------------------
__global__ void k_gate_add(const float* __restrict__ pf, const float* __restrict__ fd,
                           const float* __restrict__ pol, float* __restrict__ out)
{
  int idx = blockIdx.x * blockDim.x + threadIdx.x;
  if (idx >= Bt * Dm) return;
  out[idx] = pf[idx] + fd[idx] * pol[idx / Dm];
}

// ------------- apply_deltas + policy blend ---------------------------------
__global__ void k_final_boxes(const float* __restrict__ deltas, const float* __restrict__ bboxes,
                              const float* __restrict__ ap, float* __restrict__ out)
{
  int i = blockIdx.x * blockDim.x + threadIdx.x;
  if (i >= Bt) return;
  float x0 = bboxes[i * 4 + 0], y0 = bboxes[i * 4 + 1];
  float x1 = bboxes[i * 4 + 2], y1 = bboxes[i * 4 + 3];
  float w = x1 - x0, h = y1 - y0;
  float cx = x0 + 0.5f * w, cy = y0 + 0.5f * h;
  float dx = deltas[i * 4 + 0] * 0.5f, dyv = deltas[i * 4 + 1] * 0.5f;
  float dw = fminf(deltas[i * 4 + 2], SCALE_CLAMP);
  float dh = fminf(deltas[i * 4 + 3], SCALE_CLAMP);
  float pcx = dx * w + cx, pcy = dyv * h + cy;
  float pw = expf(dw) * w, ph = expf(dh) * h;
  float a = ap[i];
  float b0 = pcx - 0.5f * pw, b1 = pcy - 0.5f * ph;
  float b2 = pcx + 0.5f * pw, b3 = pcy + 0.5f * ph;
  out[i * 4 + 0] = x0 * (1.f - a) + b0 * a;
  out[i * 4 + 1] = y0 * (1.f - a) + b1 * a;
  out[i * 4 + 2] = x1 * (1.f - a) + b2 * a;
  out[i * 4 + 3] = y1 * (1.f - a) + b3 * a;
}

__global__ void k_final_logits(const float* __restrict__ logits, const float* __restrict__ cl,
                               const float* __restrict__ ap, float* __restrict__ out)
{
  int idx = blockIdx.x * blockDim.x + threadIdx.x;
  if (idx >= Bt * NC) return;
  float a = ap[idx / NC];
  out[idx] = logits[idx] * (1.f - a) + cl[idx] * a;
}

// ---------------------------------------------------------------------------
static inline dim3 gemm_grid(int M, int N, int batch) {
  return dim3((N + 63) / 64, (M + 15) / 16, batch);
}

#define GEMM(ACT, TB, A, Bm, bias, C, M, N, K, lda, ldb, ldc, bsA, bsB, bsC, alpha, batch) \
  k_gemm<ACT, TB><<<gemm_grid(M, N, batch), 128, 0, stream>>>( \
      A, Bm, bias, C, M, N, K, lda, ldb, ldc, (long long)(bsA), (long long)(bsB), (long long)(bsC), alpha)

struct MhaBufs { float *qkv, *Q, *K, *V, *S, *O, *merged; };

// Enqueue a full 8-head MHA over (B,256); result (with proj bias) -> outp.
static void run_mha(hipStream_t stream, const float* x,
                    const float* qkv_w, const float* qkv_b,
                    const float* proj_w, const float* proj_b,
                    const MhaBufs& w, float* outp)
{
  GEMM(0, 0, x, qkv_w, qkv_b, w.qkv, Bt, 3 * Dm, Dm, Dm, 3 * Dm, 3 * Dm, 0, 0, 0, 1.f, 1);
  k_qkv_split<<<(Bt * Dm + 255) / 256, 256, 0, stream>>>(w.qkv, w.Q, w.K, w.V);
  // scores = Q @ K^T * hd^-0.5   (NH batches of 300x300, K=32) -- B transposed
  GEMM(0, 1, w.Q, w.K, (const float*)nullptr, w.S, NR, NR, HD, HD, HD, NR,
       NR * HD, NR * HD, NR * NR, 0.17677669529663687f, NH);
  k_softmax300<<<NH * NR, 128, 0, stream>>>(w.S);
  // O = softmax(scores) @ V     (NH batches of 300x32, K=300)
  GEMM(0, 0, w.S, w.V, (const float*)nullptr, w.O, NR, HD, NR, NR, HD, HD,
       NR * NR, NR * HD, NR * HD, 1.f, NH);
  k_head_merge<<<(Bt * Dm + 255) / 256, 256, 0, stream>>>(w.O, w.merged);
  GEMM(0, 0, w.merged, proj_w, proj_b, outp, Bt, Dm, Dm, Dm, Dm, Dm, 0, 0, 0, 1.f, 1);
}

extern "C" void kernel_launch(void* const* d_in, const int* in_sizes, int n_in,
                              void* d_out, int out_size, void* d_ws, size_t ws_size,
                              hipStream_t stream) {
  (void)in_sizes; (void)n_in; (void)out_size; (void)ws_size;
  // --- inputs (top level in setup_inputs() order; the nested `params` dict is
  //     assumed flattened JAX-pytree style, i.e. sorted keys at every level) ---
  const float* pro    = (const float*)d_in[0];   // (8,300,256)
  const float* roi    = (const float*)d_in[1];   // (49,2400,256)
  const float* logits = (const float*)d_in[2];   // (2400,80)
  const float* bboxes = (const float*)d_in[3];   // (2400,4)
  const float* policy = (const float*)d_in[4];   // (2400)
  const float* pol_st = (const float*)d_in[5];   // (2400)
  const float* attn_proj_b = (const float*)d_in[6];
  const float* attn_proj_w = (const float*)d_in[7];
  const float* attn_qkv_b  = (const float*)d_in[8];
  const float* attn_qkv_w  = (const float*)d_in[9];
  const float* box_b   = (const float*)d_in[10];
  const float* box_w   = (const float*)d_in[11];
  const float* cls_lnb = (const float*)d_in[12];
  const float* cls_lng = (const float*)d_in[13];
  const float* cls_w   = (const float*)d_in[14];
  const float* clso_b  = (const float*)d_in[15];
  const float* clso_w  = (const float*)d_in[16];
  const float* dyn_b    = (const float*)d_in[17];
  const float* dyn_ln1b = (const float*)d_in[18];
  const float* dyn_ln1g = (const float*)d_in[19];
  const float* dyn_ln2b = (const float*)d_in[20];
  const float* dyn_ln2g = (const float*)d_in[21];
  const float* dyn_ln3b = (const float*)d_in[22];
  const float* dyn_ln3g = (const float*)d_in[23];
  const float* dyn_outb = (const float*)d_in[24];
  const float* dyn_outw = (const float*)d_in[25];
  const float* dyn_w    = (const float*)d_in[26];
  const float* ffn_b1 = (const float*)d_in[27];
  const float* ffn_b2 = (const float*)d_in[28];
  const float* ffn_w1 = (const float*)d_in[29];
  const float* ffn_w2 = (const float*)d_in[30];
  const float* n1b = (const float*)d_in[31];
  const float* n1g = (const float*)d_in[32];
  const float* n2b = (const float*)d_in[33];
  const float* n2g = (const float*)d_in[34];
  const float* n3b = (const float*)d_in[35];
  const float* n3g = (const float*)d_in[36];
  const float* p_proj_b = (const float*)d_in[37];
  const float* p_proj_w = (const float*)d_in[38];
  const float* p_qkv_b  = (const float*)d_in[39];
  const float* p_qkv_w  = (const float*)d_in[40];
  const float* p_b1  = (const float*)d_in[41];
  const float* p_b2  = (const float*)d_in[42];
  const float* p_dyb = (const float*)d_in[43];
  const float* p_dyw = (const float*)d_in[44];
  const float* p_lnb = (const float*)d_in[45];
  const float* p_lng = (const float*)d_in[46];
  const float* p_stb = (const float*)d_in[47];
  const float* p_stw = (const float*)d_in[48];
  const float* p_w1  = (const float*)d_in[49];
  const float* p_w2  = (const float*)d_in[50];
  const float* reg_lnb = (const float*)d_in[51];
  const float* reg_lng = (const float*)d_in[52];
  const float* reg_w   = (const float*)d_in[53];

  float* out_logits = (float*)d_out;             // (2400,80)
  float* out_boxes  = (float*)d_out + Bt * NC;   // (2400,4)

  // --- workspace bump allocator (~552 MB of f32; dominated by `pr` 315 MB) ---
  size_t off = 0;
  auto alloc = [&](size_t elems) { float* p = (float*)d_ws + off; off += elems; return p; };
  MhaBufs mw;
  mw.qkv    = alloc((size_t)Bt * 3 * Dm);
  mw.Q      = alloc((size_t)NH * NR * HD);
  mw.K      = alloc((size_t)NH * NR * HD);
  mw.V      = alloc((size_t)NH * NR * HD);
  mw.S      = alloc((size_t)NH * NR * NR);
  mw.O      = alloc((size_t)NH * NR * HD);
  mw.merged = alloc((size_t)Bt * Dm);
  float* t_mha = alloc((size_t)Bt * Dm);
  float* xP    = alloc((size_t)Bt * Dm);
  float* h1    = alloc((size_t)Bt * (Dm / 2));
  float* h2    = alloc((size_t)Bt * (Dm / 4));
  float* polg  = alloc(Bt);
  float* ap    = alloc(Bt);
  float* pf    = alloc((size_t)Bt * Dm);
  float* pr    = alloc((size_t)Bt * 2 * Dm * DDY);    // generated dyn params
  float* f1    = alloc((size_t)Bt * Pp * DDY);
  float* f2    = alloc((size_t)Bt * Pp * Dm);
  float* fdyn  = alloc((size_t)Bt * Dm);
  float* pf3   = alloc((size_t)Bt * Dm);
  float* obj   = alloc((size_t)Bt * Dm);
  float* ffnh  = alloc((size_t)Bt * DFF);
  float* ffno  = alloc((size_t)Bt * Dm);
  float* obj2  = alloc((size_t)Bt * Dm);
  float* clsT  = alloc((size_t)Bt * Dm);
  float* regT  = alloc((size_t)Bt * Dm);
  float* clraw = alloc((size_t)Bt * NC);
  float* delt  = alloc((size_t)Bt * 4);

  // ================= prune-score predictor branch =================
  run_mha(stream, pro, p_qkv_w, p_qkv_b, p_proj_w, p_proj_b, mw, t_mha);
  k_layernorm<Dm><<<Bt, 64, 0, stream>>>(pro, t_mha, p_lng, p_lnb, xP, 0);
  GEMM(2, 0, xP, p_w1, p_b1, h1, Bt, Dm / 2, Dm, Dm, Dm / 2, Dm / 2, 0, 0, 0, 1.f, 1);   // gelu
  GEMM(2, 0, h1, p_w2, p_b2, h2, Bt, Dm / 4, Dm / 2, Dm / 2, Dm / 4, Dm / 4, 0, 0, 0, 1.f, 1);
  k_policy<<<(Bt + 127) / 128, 128, 0, stream>>>(h2, p_dyw, p_dyb, p_stw, p_stb,
                                                 policy, pol_st, polg, ap);

  // ================= main self-attention =================
  run_mha(stream, pro, attn_qkv_w, attn_qkv_b, attn_proj_w, attn_proj_b, mw, t_mha);
  k_layernorm<Dm><<<Bt, 64, 0, stream>>>(pro, t_mha, n1g, n1b, pf, 0);

  // ================= DynamicConv =================
  // pr = pf @ dyn_w + dyn_b : (2400, 32768)  -- the 40-GFLOP matmul
  GEMM(0, 0, pf, dyn_w, dyn_b, pr, Bt, 2 * Dm * DDY, Dm, Dm, 2 * Dm * DDY, 2 * Dm * DDY,
       0, 0, 0, 1.f, 1);
  // f1[b] = feats[b] (49x256, read straight from roi with lda=B*D) @ p1[b] (256x64)
  GEMM(0, 0, roi, pr, (const float*)nullptr, f1, Pp, DDY, Dm,
       Bt * Dm, DDY, DDY, Dm, 2 * Dm * DDY, Pp * DDY, 1.f, Bt);
  k_layernorm<DDY><<<Bt * Pp, 64, 0, stream>>>(f1, (const float*)nullptr,
                                               dyn_ln1g, dyn_ln1b, f1, 1);
  // f2[b] = f1[b] (49x64) @ p2[b] (64x256)
  GEMM(0, 0, f1, pr + (size_t)Dm * DDY, (const float*)nullptr, f2, Pp, Dm, DDY,
       DDY, Dm, Dm, Pp * DDY, 2 * Dm * DDY, Pp * Dm, 1.f, Bt);
  k_layernorm<Dm><<<Bt * Pp, 64, 0, stream>>>(f2, (const float*)nullptr,
                                              dyn_ln2g, dyn_ln2b, f2, 1);
  // fdyn = f2.reshape(B, P*D) @ out_w + out_b  (K = 12544)
  GEMM(0, 0, f2, dyn_outw, dyn_outb, fdyn, Bt, Dm, Pp * Dm, Pp * Dm, Dm, Dm,
       0, 0, 0, 1.f, 1);
  k_layernorm<Dm><<<Bt, 64, 0, stream>>>(fdyn, (const float*)nullptr,
                                         dyn_ln3g, dyn_ln3b, fdyn, 1);
  k_gate_add<<<(Bt * Dm + 255) / 256, 256, 0, stream>>>(pf, fdyn, polg, pf3);
  k_layernorm<Dm><<<Bt, 64, 0, stream>>>(pf3, (const float*)nullptr, n2g, n2b, obj, 0);

  // ================= FFN =================
  GEMM(1, 0, obj, ffn_w1, ffn_b1, ffnh, Bt, DFF, Dm, Dm, DFF, DFF, 0, 0, 0, 1.f, 1); // relu
  GEMM(0, 0, ffnh, ffn_w2, ffn_b2, ffno, Bt, Dm, DFF, DFF, Dm, Dm, 0, 0, 0, 1.f, 1);
  k_layernorm<Dm><<<Bt, 64, 0, stream>>>(obj, ffno, n3g, n3b, obj2, 0);

  // ================= heads =================
  GEMM(0, 0, obj2, cls_w, (const float*)nullptr, clsT, Bt, Dm, Dm, Dm, Dm, Dm, 0, 0, 0, 1.f, 1);
  k_layernorm<Dm><<<Bt, 64, 0, stream>>>(clsT, (const float*)nullptr, cls_lng, cls_lnb, clsT, 1);
  GEMM(0, 0, obj2, reg_w, (const float*)nullptr, regT, Bt, Dm, Dm, Dm, Dm, Dm, 0, 0, 0, 1.f, 1);
  k_layernorm<Dm><<<Bt, 64, 0, stream>>>(regT, (const float*)nullptr, reg_lng, reg_lnb, regT, 1);
  GEMM(0, 0, clsT, clso_w, clso_b, clraw, Bt, NC, Dm, Dm, NC, NC, 0, 0, 0, 1.f, 1);
  GEMM(0, 0, regT, box_w, box_b, delt, Bt, 4, Dm, Dm, 4, 4, 0, 0, 0, 1.f, 1);

  // ================= blend & output =================
  k_final_logits<<<(Bt * NC + 255) / 256, 256, 0, stream>>>(logits, clraw, ap, out_logits);
  k_final_boxes<<<(Bt + 127) / 128, 128, 0, stream>>>(delt, bboxes, ap, out_boxes);
}
